// MultiheadAttention_32950989095606
// MI455X (gfx1250) — matrix operand order repaired and checked by reading, not compiled, not run on hardware
//
#include <hip/hip_runtime.h>
#include <hip/hip_bf16.h>

typedef _Float16 v16h __attribute__((ext_vector_type(16)));
typedef float    v8f  __attribute__((ext_vector_type(8)));

#define E_      768
#define H_      12
#define D_      64
#define G_      8
#define S_      197
#define B_      64
#define MAXS_   197
#define SP_     224          // seq padded to multiple of 32
#define KAUG_   128          // augmented K: [q|rel_s] x [k+rel_t|k]
#define SCALING 0.125f       // D^-0.5

// ---- WMMA fragment helpers (CDNA5 ISA 7.12.2, wave32) ----
// A 16x32 f16: lane&15 = M ; half j -> K = (lane>>4)*16 + j (contiguous).
__device__ __forceinline__ int akidx(int j, int hi) {
  return j + hi * 16;
}
// Inverse mapping: store value for (row M, k) where the A-frag load will find it.
// Layout: swz[(kc*32 + lane)*16 + j], fragment load = one aligned v16h per lane.
__device__ __forceinline__ void a_swz_store(_Float16* swz, int kglob, int Mrow,
                                            _Float16 v) {
  const int kc = kglob >> 5;
  const int kk = kglob & 31;
  const int hi = (kk >> 4) & 1;
  const int j = kk - hi * 16;                   // {0..15}
  swz[((kc * 32) + (Mrow + hi * 16)) * 16 + j] = v;
}
// B 32x16 f16: lane&15 = N ; half j -> K = (lane>>4)*16 + j (contiguous v16h)
// C/D f32 16x16: element r -> (M = r + (lane>>4)*8, N = lane&15)

__device__ __forceinline__ v8f wmma32(v16h a, v16h b, v8f c) {
  return __builtin_amdgcn_wmma_f32_16x16x32_f16(false, a, false, b,
                                                (short)0, c, false, false);
}

// =====================================================================
// Kernel P: one-time f16 weight prep.
//   qkvH[2304][96]  = f16(qkv_w)                       (B-frag friendly)
//   fc1T[96][768]   = f16(fc1_w^T)                     (B[k=i][n=o])
//   fc2T[768][96]   = f16(fc2_w^T)                     (B[k=o][n=e])
//   outH[768][96]   = f16(out_w)                       (already [n][k])
// =====================================================================
__global__ __launch_bounds__(256)
void wprep_kernel(const float* __restrict__ qkv_w, const float* __restrict__ fc1_w,
                  const float* __restrict__ fc2_w, const float* __restrict__ out_w,
                  _Float16* __restrict__ qkvH, _Float16* __restrict__ fc1T,
                  _Float16* __restrict__ fc2T, _Float16* __restrict__ outH) {
  const int idx = blockIdx.x * 256 + threadIdx.x;
  if (idx < 2304 * 96) qkvH[idx] = (_Float16)qkv_w[idx];
  if (idx < 96 * 768) {
    const int o = idx / 768, i = idx - o * 768;
    fc1T[idx] = (_Float16)fc1_w[(size_t)i * 96 + o];
  }
  if (idx < 768 * 96) {
    const int e = idx / 96, i = idx - e * 96;
    fc2T[idx] = (_Float16)fc2_w[(size_t)i * 768 + e];
    outH[idx] = (_Float16)out_w[idx];
  }
}

// =====================================================================
// Kernel 0: grouped QKV GEMM (8 groups, M=12608, K=96, N=288) via WMMA.
// Epilogue: +bias, channel shuffle (i=o*8+g), split q/k/v, q*=scaling:
//   QA[bh][s][0:64]=q*scale ; KB[bh][t][64:128]=k ; Vt[bh][d][t]=v
// =====================================================================
__global__ __launch_bounds__(128)
void qkv_gemm_kernel(const float* __restrict__ query,
                     const _Float16* __restrict__ qkvH,
                     const float* __restrict__ qkv_b,
                     _Float16* __restrict__ QA,
                     _Float16* __restrict__ KB,
                     _Float16* __restrict__ Vt) {
  const int wave = threadIdx.x >> 5;
  const int lane = threadIdx.x & 31;
  const int lo = lane & 15, hi = lane >> 4;
  const int ntile = blockIdx.y * 4 + wave;          // 0..143 = 8 groups * 18 tiles
  const int g  = ntile / 18;
  const int n0 = (ntile % 18) * 16;
  const int m0 = blockIdx.x * 16;                   // 788 M-tiles

  const int mA = m0 + lo;
  const int bA = mA / S_;
  const int sA = mA - bA * S_;
  const float* qrow = query + ((size_t)sA * B_ + bA) * E_ + g * 96;
  const _Float16* wrow = qkvH + (size_t)(g * 288 + n0 + lo) * 96;

  v8f c = {};
#pragma unroll
  for (int kc = 0; kc < 3; ++kc) {                  // K = 96 = 3 x 32
    v16h a;
#pragma unroll
    for (int j = 0; j < 16; ++j)
      a[j] = (_Float16)qrow[kc * 32 + akidx(j, hi)];
    const v16h bfr = *(const v16h*)(wrow + kc * 32 + hi * 16);
    c = wmma32(a, bfr, c);
  }

  const int o = n0 + lo;
  const float bias = qkv_b[g * 288 + o];
  const int i = o * 8 + g;                          // shuffled channel

#pragma unroll
  for (int r = 0; r < 8; ++r) {
    const int m = m0 + r + hi * 8;
    const int b_ = m / S_;
    const int s  = m - b_ * S_;
    const float val = c[r] + bias;
    if (i < E_) {
      const int h = i >> 6, d = i & 63;
      QA[((size_t)(b_ * H_ + h) * SP_ + s) * KAUG_ + d] = (_Float16)(val * SCALING);
    } else if (i < 2 * E_) {
      const int e = i - E_;
      const int h = e >> 6, d = e & 63;
      KB[((size_t)(b_ * H_ + h) * SP_ + s) * KAUG_ + 64 + d] = (_Float16)val;
    } else {
      const int e = i - 2 * E_;
      const int h = e >> 6, d = e & 63;
      Vt[((size_t)(b_ * H_ + h) * D_ + d) * SP_ + s] = (_Float16)val;
    }
  }
}

// =====================================================================
// Kernel 1: rel-pos injection + zero seq padding.
// =====================================================================
__global__ __launch_bounds__(256)
void fixup_kernel(const float* __restrict__ rel_pos,
                  _Float16* __restrict__ QA,
                  _Float16* __restrict__ KB,
                  _Float16* __restrict__ Vt) {
  const size_t bh = blockIdx.x;
  for (int idx = threadIdx.x; idx < SP_ * 64; idx += 256) {
    const int s = idx >> 6, d = idx & 63;
    _Float16* qa = QA + (bh * SP_ + s) * KAUG_;
    _Float16* kb = KB + (bh * SP_ + s) * KAUG_;
    if (s < S_) {
      const float r = rel_pos[(size_t)(MAXS_ - 1 + s) * D_ + d];
      qa[64 + d] = (_Float16)r;
      kb[d] = (_Float16)((float)kb[64 + d] + r);
    } else {
      qa[d] = (_Float16)0.f; qa[64 + d] = (_Float16)0.f;
      kb[d] = (_Float16)0.f; kb[64 + d] = (_Float16)0.f;
      Vt[(bh * D_ + d) * SP_ + s] = (_Float16)0.f;
    }
  }
}

// =====================================================================
// Kernel 2: attention. One WG (4 waves) per (batch, 16-row s-block),
// looping 12 heads. QA tile and probs are written once into pre-swizzled
// f16 LDS A-layouts so every A-fragment is a single aligned v16h LDS load.
// =====================================================================
__global__ __launch_bounds__(128)
void attn_kernel(const _Float16* __restrict__ QA,
                 const _Float16* __restrict__ KB,
                 const _Float16* __restrict__ Vt,
                 float* __restrict__ ctx,
                 float* __restrict__ avg_out) {
  __shared__ float sc[16][SP_];                       // raw scores
  __shared__ float avgw[16][SP_];                     // sum_h probs / H
  __shared__ float redm[16][8];
  __shared__ float reds[16][8];
  __shared__ alignas(32) _Float16 qsw[4 * 32 * 16];   // swizzled QA tile (K=128)
  __shared__ alignas(32) _Float16 psw[(SP_ / 32) * 32 * 16]; // swizzled probs

  const int tid = threadIdx.x;
  const int wave = tid >> 5, lane = tid & 31;
  const int lo = lane & 15, hi = lane >> 4;
  const int s0 = blockIdx.x * 16;
  const int b  = blockIdx.y;

  for (int idx = tid; idx < 16 * SP_; idx += 128) ((float*)avgw)[idx] = 0.f;
  __syncthreads();

  const int row = tid >> 3, sub = tid & 7;

  for (int h = 0; h < H_; ++h) {
    const size_t bh = (size_t)b * H_ + h;

    // ---- stage QA rows (coalesced) into swizzled A-layout ----
    for (int idx = tid; idx < 16 * KAUG_; idx += 128) {
      const int r = idx >> 7, k = idx & 127;
      a_swz_store(qsw, k, r, QA[(bh * SP_ + (size_t)(s0 + r)) * KAUG_ + k]);
    }
    __syncthreads();

    v16h aq[4];
#pragma unroll
    for (int kc = 0; kc < 4; ++kc)
      aq[kc] = *(const v16h*)(qsw + ((kc * 32) + lane) * 16);

    // ---- scores: 14 column tiles, round-robin over waves ----
    for (int jt = wave; jt < SP_ / 16; jt += 4) {
      v8f c = {};
#pragma unroll
      for (int kc = 0; kc < 4; ++kc) {
        const v16h bfr = *(const v16h*)(KB + (bh * SP_ + (size_t)(jt * 16 + lo)) * KAUG_
                                        + kc * 32 + hi * 16);
        c = wmma32(aq[kc], bfr, c);
      }
#pragma unroll
      for (int r = 0; r < 8; ++r)
        sc[r + hi * 8][jt * 16 + lo] = c[r];
    }
    __syncthreads();

    // ---- masked softmax; probs -> swizzled f16; accumulate avg ----
    float mx = -3.0e38f;
    for (int t = sub; t < S_; t += 8) mx = fmaxf(mx, sc[row][t]);
    redm[row][sub] = mx;
    __syncthreads();
    const float rmax =
        fmaxf(fmaxf(fmaxf(redm[row][0], redm[row][1]), fmaxf(redm[row][2], redm[row][3])),
              fmaxf(fmaxf(redm[row][4], redm[row][5]), fmaxf(redm[row][6], redm[row][7])));
    float psum = 0.f;
    for (int t = sub; t < SP_; t += 8) {
      const float e = (t < S_) ? __expf(sc[row][t] - rmax) : 0.f;
      sc[row][t] = e;
      psum += e;
    }
    reds[row][sub] = psum;
    __syncthreads();
    const float rsum = reds[row][0] + reds[row][1] + reds[row][2] + reds[row][3] +
                       reds[row][4] + reds[row][5] + reds[row][6] + reds[row][7];
    const float inv = 1.f / rsum;
    for (int t = sub; t < SP_; t += 8) {
      const float p = sc[row][t] * inv;
      a_swz_store(psw, t, row, (_Float16)p);
      avgw[row][t] += p * (1.f / (float)H_);
    }
    __syncthreads();

    // ---- out = probs @ V : each wave owns one 16-wide d-tile ----
    {
      const int d0 = wave * 16;
      v8f c = {};
#pragma unroll
      for (int kc = 0; kc < SP_ / 32; ++kc) {
        const v16h ap  = *(const v16h*)(psw + ((kc * 32) + lane) * 16);
        const v16h bfr = *(const v16h*)(Vt + (bh * D_ + (size_t)(d0 + lo)) * SP_
                                        + kc * 32 + hi * 16);
        c = wmma32(ap, bfr, c);
      }
#pragma unroll
      for (int r = 0; r < 8; ++r) {
        const int s = s0 + r + hi * 8;
        if (s < S_)
          ctx[((size_t)b * S_ + s) * E_ + h * D_ + d0 + lo] = c[r];
      }
    }
    __syncthreads();
  }

  for (int idx = tid; idx < 16 * SP_; idx += 128) {
    const int rr = idx / SP_, t = idx % SP_;
    const int s = s0 + rr;
    if (s < S_ && t < S_)
      avg_out[((size_t)b * S_ + s) * S_ + t] = avgw[rr][t];
  }
}

// =====================================================================
// Kernel 3: 16-row tile, fully tensorized epilogue:
//   LN (VALU reductions) -> fc1 WMMA (K=768) + gelu -> fc2 WMMA (K=96)
//   + sigmoid gate -> grouped out-proj WMMA (K=96) -> write (S,B,E).
// =====================================================================
__global__ __launch_bounds__(128)
void mlp_out_kernel(const float* __restrict__ ctx,
                    const float* __restrict__ ln_g, const float* __restrict__ ln_b,
                    const _Float16* __restrict__ fc1T, const float* __restrict__ fc1_b,
                    const _Float16* __restrict__ fc2T, const float* __restrict__ fc2_b,
                    const _Float16* __restrict__ outH, const float* __restrict__ out_b,
                    float* __restrict__ out0) {
  __shared__ float rowv[16][E_];                     // 48 KB raw ctx rows
  __shared__ alignas(32) _Float16 asw[24 * 32 * 16]; // swizzled A (ln, then gated)
  __shared__ alignas(32) _Float16 y1sw[3 * 32 * 16]; // swizzled gelu(fc1)
  __shared__ float redb[16][8];
  __shared__ float mu16[16], rs16[16];

  const int tid = threadIdx.x;
  const int wave = tid >> 5, lane = tid & 31;
  const int lo = lane & 15, hi = lane >> 4;
  const int m0 = blockIdx.x * 16;                    // 788 tiles, no remainder

  for (int r = 0; r < 16; ++r)
    for (int e = tid; e < E_; e += 128)
      rowv[r][e] = ctx[(size_t)(m0 + r) * E_ + e];
  __syncthreads();

  // ---- per-row mean / rstd (8 threads per row) ----
  const int rw = tid >> 3, sub = tid & 7;
  float ps = 0.f;
  for (int e = sub; e < E_; e += 8) ps += rowv[rw][e];
  redb[rw][sub] = ps;
  __syncthreads();
  if (tid < 16) {
    float s = 0.f;
    for (int u = 0; u < 8; ++u) s += redb[tid][u];
    mu16[tid] = s * (1.f / E_);
  }
  __syncthreads();
  const float mu = mu16[rw];
  float pv = 0.f;
  for (int e = sub; e < E_; e += 8) { const float d = rowv[rw][e] - mu; pv += d * d; }
  redb[rw][sub] = pv;
  __syncthreads();
  if (tid < 16) {
    float s = 0.f;
    for (int u = 0; u < 8; ++u) s += redb[tid][u];
    rs16[tid] = rsqrtf(s * (1.f / E_) + 1e-5f);
  }
  __syncthreads();

  // ---- LN -> swizzled f16 A ----
  for (int r = 0; r < 16; ++r) {
    const float m_ = mu16[r], rs_ = rs16[r];
    for (int e = tid; e < E_; e += 128)
      a_swz_store(asw, e, r,
                  (_Float16)((rowv[r][e] - m_) * rs_ * ln_g[e] + ln_b[e]));
  }
  __syncthreads();

  // ---- fc1: [16x768]x[768x96] + gelu -> y1sw ----
  for (int jt = wave; jt < 6; jt += 4) {
    const int n = jt * 16 + lo;
    const float bias = fc1_b[n];
    v8f c = {};
    for (int kc = 0; kc < 24; ++kc) {
      const v16h a   = *(const v16h*)(asw + ((kc * 32) + lane) * 16);
      const v16h bfr = *(const v16h*)(fc1T + (size_t)n * E_ + kc * 32 + hi * 16);
      c = wmma32(a, bfr, c);
    }
#pragma unroll
    for (int r = 0; r < 8; ++r) {
      const float x = c[r] + bias;
      const float t = 0.7978845608028654f * (x + 0.044715f * x * x * x);
      a_swz_store(y1sw, n, r + hi * 8, (_Float16)(0.5f * x * (1.f + tanhf(t))));
    }
  }
  __syncthreads();

  // ---- fc2: [16x96]x[96x768] + sigmoid gate -> gated rows into asw ----
  for (int jt = wave; jt < 48; jt += 4) {
    const int n = jt * 16 + lo;
    const float bias = fc2_b[n];
    v8f c = {};
#pragma unroll
    for (int kc = 0; kc < 3; ++kc) {
      const v16h a   = *(const v16h*)(y1sw + ((kc * 32) + lane) * 16);
      const v16h bfr = *(const v16h*)(fc2T + (size_t)n * 96 + kc * 32 + hi * 16);
      c = wmma32(a, bfr, c);
    }
#pragma unroll
    for (int r = 0; r < 8; ++r) {
      const int M = r + hi * 8;
      const float sg = 1.f / (1.f + __expf(-(c[r] + bias)));
      a_swz_store(asw, n, M, (_Float16)(rowv[M][n] * sg));
    }
  }
  __syncthreads();

  // ---- grouped out-proj: per group g, [16x96]x[96x96] -> out (S,B,E) ----
  for (int jt = wave; jt < 48; jt += 4) {
    const int eo = jt * 16 + lo;                    // tile fits inside one group
    const int g = jt / 6;
    const float bias = out_b[eo];
    v8f c = {};
#pragma unroll
    for (int kc = 0; kc < 3; ++kc) {
      const v16h a   = *(const v16h*)(asw + (((3 * g + kc) * 32) + lane) * 16);
      const v16h bfr = *(const v16h*)(outH + (size_t)eo * 96 + kc * 32 + hi * 16);
      c = wmma32(a, bfr, c);
    }
#pragma unroll
    for (int r = 0; r < 8; ++r) {
      const int m = m0 + r + hi * 8;
      const int b_ = m / S_;
      const int s_ = m - b_ * S_;
      out0[((size_t)s_ * B_ + b_) * E_ + eo] = c[r] + bias;
    }
  }
}

// =====================================================================
extern "C" void kernel_launch(void* const* d_in, const int* in_sizes, int n_in,
                              void* d_out, int out_size, void* d_ws, size_t ws_size,
                              hipStream_t stream) {
  (void)in_sizes; (void)n_in; (void)out_size; (void)ws_size;
  const float* query = (const float*)d_in[0];
  // d_in[1]=key, d_in[2]=value unused by the reference (qkv from query only)
  const float* qkv_w = (const float*)d_in[3];
  const float* qkv_b = (const float*)d_in[4];
  const float* out_w = (const float*)d_in[5];
  const float* out_b = (const float*)d_in[6];
  const float* rel   = (const float*)d_in[7];
  const float* ln_g  = (const float*)d_in[8];
  const float* ln_b  = (const float*)d_in[9];
  const float* fc1_w = (const float*)d_in[10];
  const float* fc1_b = (const float*)d_in[11];
  const float* fc2_w = (const float*)d_in[12];
  const float* fc2_b = (const float*)d_in[13];

  // workspace carve-up (all offsets 32B-aligned for v16h loads)
  const size_t qaElems = (size_t)B_ * H_ * SP_ * KAUG_;
  _Float16* QA = (_Float16*)d_ws;
  _Float16* KB = QA + qaElems;
  _Float16* Vt = KB + qaElems;
  float* ctx = (float*)(Vt + (size_t)B_ * H_ * D_ * SP_);
  _Float16* qkvH = (_Float16*)(ctx + (size_t)B_ * S_ * E_);
  _Float16* fc1T = qkvH + (size_t)2304 * 96;
  _Float16* fc2T = fc1T + (size_t)96 * E_;
  _Float16* outH = fc2T + (size_t)E_ * 96;

  float* out0 = (float*)d_out;                       // (S,B,E)
  float* avg  = out0 + (size_t)S_ * B_ * E_;         // (B,S,S)

  wprep_kernel<<<dim3((2304 * 96 + 255) / 256), 256, 0, stream>>>(
      qkv_w, fc1_w, fc2_w, out_w, qkvH, fc1T, fc2T, outH);
  qkv_gemm_kernel<<<dim3((B_ * S_) / 16, 36), 128, 0, stream>>>(
      query, qkvH, qkv_b, QA, KB, Vt);
  fixup_kernel<<<dim3(B_ * H_), 256, 0, stream>>>(rel, QA, KB, Vt);
  attn_kernel<<<dim3(13, B_), 128, 0, stream>>>(QA, KB, Vt, ctx, avg);
  mlp_out_kernel<<<dim3((B_ * S_) / 16), 128, 0, stream>>>(
      ctx, ln_g, ln_b, fc1T, fc1_b, fc2T, fc2_b, outH, out_b, out0);
}
